// Router_704374636924
// MI455X (gfx1250) — matrix operand order repaired and checked by reading, not compiled, not run on hardware
//
#include <hip/hip_runtime.h>
#include <hip/hip_bf16.h>

// ---------------------------------------------------------------------------
// MoE router for MI455X (gfx1250, wave32, WMMA).
// scores[N,E] = (x + 0.01*noise) @ W^T ; top-2 per row; expert histogram; MSE.
// HBM-bound (AI ~16 flop/B): stream x/noise once via b128 loads, W lives in
// LDS (transposed, padded), GEMM via v_wmma_f32_16x16x32_bf16.
// ---------------------------------------------------------------------------

#define N_TOK   262144
#define DIM     1024
#define NEXP    64
#define TOPK    2
#define NOISE_EPS 0.01f

#define WAVES_PER_BLOCK 8
#define TILES_PER_WAVE  4
#define NBLOCKS         512   // 512*8*4 waves-tiles * 16 tokens = 262144

// LDS: W transposed [DIM][NEXP + pad] bf16. Pad=8 -> row stride 144B = 36
// dwords; B fragments are read as single bf16s: 16 lanes take 16 adjacent
// experts of one row, the two half-waves rows 8 apart (288 dwords).
#define WT_PITCH   (NEXP + 8)                      // 72 bf16
#define WT_BYTES   (DIM * WT_PITCH * 2)            // 147456
#define SCORE_BYTES (WAVES_PER_BLOCK * 16 * NEXP * 4) // 32768
#define SMEM_BYTES (WT_BYTES + SCORE_BYTES)

typedef __attribute__((ext_vector_type(8)))  float  v8f;
typedef __attribute__((ext_vector_type(16))) __bf16 v16bf;
typedef __attribute__((ext_vector_type(8)))  __bf16 v8bf;

__global__ void router_init_counts(unsigned* counts) {
    if (threadIdx.x < NEXP) counts[threadIdx.x] = 0u;
}

__global__ __launch_bounds__(256)
void router_gemm_topk(const float* __restrict__ x,
                      const float* __restrict__ noise,
                      const float* __restrict__ W,
                      float* __restrict__ out_w,
                      int* __restrict__ out_i,
                      unsigned* __restrict__ counts) {
    extern __shared__ char smem[];
    __bf16* Wt     = reinterpret_cast<__bf16*>(smem);
    float*  scores = reinterpret_cast<float*>(smem + WT_BYTES);

    const int tid  = threadIdx.x;
    const int lane = tid & 31;
    const int wave = tid >> 5;
    const int mlan = lane & 15;     // token-in-tile (A) / expert-in-tile (C)
    const int half = lane >> 4;

    // --- Stage W^T into LDS as bf16 (coalesced global read, padded rows) ---
    for (int i = tid; i < NEXP * DIM; i += 256) {
        const int e = i >> 10;          // expert row in W[E][D]
        const int d = i & (DIM - 1);
        Wt[d * WT_PITCH + e] = (__bf16)W[i];
    }
    __syncthreads();

    float* swave = scores + wave * 16 * NEXP;

    for (int iter = 0; iter < TILES_PER_WAVE; ++iter) {
        const int tile = (blockIdx.x * WAVES_PER_BLOCK + wave) * TILES_PER_WAVE + iter;
        const int row  = tile * 16 + mlan;
        const float* xrow = x     + (size_t)row * DIM;
        const float* nrow = noise + (size_t)row * DIM;

        v8f acc[4] = {v8f{}, v8f{}, v8f{}, v8f{}};

        for (int ks = 0; ks < DIM; ks += 32) {
            // A fragment 16x32 bf16: lane holds M=mlan; VGPR v packs
            // K = 16*(v/4) + 8*half + 2*(v%4) + {0,1}  => per 16-K group the
            // lane's 8 elements are 8 *contiguous* floats: two float4 loads.
            v16bf a;
#pragma unroll
            for (int g = 0; g < 2; ++g) {
                const int k0 = ks + 16 * g + 8 * half;
                const float4 xa = *reinterpret_cast<const float4*>(xrow + k0);
                const float4 xb = *reinterpret_cast<const float4*>(xrow + k0 + 4);
                const float4 na = *reinterpret_cast<const float4*>(nrow + k0);
                const float4 nb = *reinterpret_cast<const float4*>(nrow + k0 + 4);
                a[8 * g + 0] = (__bf16)__builtin_fmaf(NOISE_EPS, na.x, xa.x);
                a[8 * g + 1] = (__bf16)__builtin_fmaf(NOISE_EPS, na.y, xa.y);
                a[8 * g + 2] = (__bf16)__builtin_fmaf(NOISE_EPS, na.z, xa.z);
                a[8 * g + 3] = (__bf16)__builtin_fmaf(NOISE_EPS, na.w, xa.w);
                a[8 * g + 4] = (__bf16)__builtin_fmaf(NOISE_EPS, nb.x, xb.x);
                a[8 * g + 5] = (__bf16)__builtin_fmaf(NOISE_EPS, nb.y, xb.y);
                a[8 * g + 6] = (__bf16)__builtin_fmaf(NOISE_EPS, nb.z, xb.z);
                a[8 * g + 7] = (__bf16)__builtin_fmaf(NOISE_EPS, nb.w, xb.w);
            }

            // B fragments 32x16 bf16: lane holds N=mlan; element s packs
            // K = 8*half + s (s<8), 16 + 8*half + (s-8) (s>=8), same as A.
            const __bf16* brow = Wt + (size_t)(ks + 8 * half) * WT_PITCH + mlan;
#pragma unroll
            for (int t = 0; t < 4; ++t) {
                v16bf b;
#pragma unroll
                for (int j = 0; j < 8; ++j) { b[j] = brow[j * WT_PITCH + t * 16]; b[8 + j] = brow[(16 + j) * WT_PITCH + t * 16]; }
                acc[t] = __builtin_amdgcn_wmma_f32_16x16x32_bf16(
                    false, a, false, b, (short)0, acc[t], false, false);
            }
        }

        // --- Stage 16x64 score tile: C layout lane: n=mlan, VGPR r -> m=r+8*half
#pragma unroll
        for (int t = 0; t < 4; ++t)
#pragma unroll
            for (int r = 0; r < 8; ++r)
                swave[(r + 8 * half) * NEXP + t * 16 + mlan] = acc[t][r];
        __syncthreads();

        // --- Top-2 per token (lanes 0..15 handle one token each) ---
        if (lane < 16) {
            const float* srow = swave + lane * NEXP;
            float b1 = -3.402823466e38f, b2 = -3.402823466e38f;
            int i1 = 0, i2 = 0;
            for (int j = 0; j < NEXP; ++j) {
                const float s = srow[j];
                if (s > b1)      { b2 = b1; i2 = i1; b1 = s; i1 = j; }
                else if (s > b2) { b2 = s; i2 = j; }
            }
            const int g = tile * 16 + lane;
            out_w[2 * g + 0] = b1;
            out_w[2 * g + 1] = b2;
            out_i[2 * g + 0] = i1;
            out_i[2 * g + 1] = i2;
            atomicAdd(&counts[i1], 1u);
            atomicAdd(&counts[i2], 1u);
        }
        __syncthreads();   // protect score buffer before next tile
    }
}

__global__ void router_aux_loss(const unsigned* __restrict__ counts,
                                float* __restrict__ aux) {
    __shared__ float sq[NEXP];
    const int t = threadIdx.x;
    const float target = (float)N_TOK * TOPK / NEXP;   // 8192
    const float d = (float)counts[t] - target;
    sq[t] = d * d;
    __syncthreads();
    if (t == 0) {
        float acc = 0.f;
        for (int i = 0; i < NEXP; ++i) acc += sq[i];
        *aux = acc * (1.0f / NEXP);
    }
}

extern "C" void kernel_launch(void* const* d_in, const int* in_sizes, int n_in,
                              void* d_out, int out_size, void* d_ws, size_t ws_size,
                              hipStream_t stream) {
    const float* x     = (const float*)d_in[0];
    const float* noise = (const float*)d_in[1];
    const float* W     = (const float*)d_in[2];

    float*  out_w = (float*)d_out;                       // [N, K] weights
    int*    out_i = (int*)((float*)d_out + 2 * N_TOK);   // [N, K] indices
    float*  aux   = (float*)d_out + 4 * N_TOK;           // scalar
    unsigned* counts = (unsigned*)d_ws;                  // [E] histogram

    router_init_counts<<<1, 64, 0, stream>>>(counts);
    router_gemm_topk<<<NBLOCKS, 256, SMEM_BYTES, stream>>>(
        x, noise, W, out_w, out_i, counts);
    router_aux_loss<<<1, NEXP, 0, stream>>>(counts, aux);
}
